// DownLayer_62182536512197
// MI455X (gfx1250) — compile-verified, matrix operand order checked by reading, not run on hardware
//
#include <hip/hip_runtime.h>
#include <hip/hip_bf16.h>

// ---------------- problem constants (from reference setup_inputs) ----------
#define B_     8
#define N_TOK  3136
#define N0_    12544
#define CIN    64
#define COUT   128
#define HH     112
#define WW     112
#define H2     56
#define W2     56
#define NS     784          // ceil(3136*0.25)
#define KCONV  576          // 9*64
#define EPSF   1e-6f

// ---------------- CDNA5 feature detection ----------------------------------
#if defined(__has_builtin)
# if __has_builtin(__builtin_amdgcn_global_load_async_to_lds_b128)
#  define HAVE_ASYNC_CP 1
# endif
# if __has_builtin(__builtin_amdgcn_s_wait_asynccnt)
#  define HAVE_ASYNC_WAIT 1
# endif
# if __has_builtin(__builtin_amdgcn_tensor_load_to_lds) && __has_builtin(__builtin_amdgcn_s_wait_tensorcnt)
#  define HAVE_TDM 1
# endif
#endif

// ---------------- WMMA / vector types (CDNA5, wave32) ----------------------
typedef __attribute__((ext_vector_type(16))) __bf16 v16bf;
typedef __attribute__((ext_vector_type(8)))  float  v8f;
typedef __attribute__((ext_vector_type(4)))  unsigned v4u;
typedef __attribute__((ext_vector_type(8)))  int      v8i;
typedef __attribute__((ext_vector_type(4)))  int      v4i;

union ABf { v16bf v; unsigned u[8]; };
union CFr { v8f   v; float    f[8]; };

__device__ inline unsigned short f2bf(float f) {
    unsigned u = __float_as_uint(f);
    unsigned r = (u + 0x7FFFu + ((u >> 16) & 1u)) >> 16;   // round-to-nearest-even
    return (unsigned short)r;
}
__device__ inline unsigned packbf(float lo, float hi) {
    return (unsigned)f2bf(lo) | ((unsigned)f2bf(hi) << 16);
}

// k-pair index (0..15) held by fragment dword v for this lane-half
__device__ inline int kpair_of(int v, int half) {
    return (v < 4) ? (v + 4 * half) : (8 + (v - 4) + 4 * half);
}

// 16B async copy global->LDS (falls back to a synchronous copy).
// Builtin params are pointers to 128-bit int vectors (per hipcc diagnostic).
__device__ inline void async_cp16(const void* g, void* l) {
#if defined(HAVE_ASYNC_CP)
    __builtin_amdgcn_global_load_async_to_lds_b128(
        (__attribute__((address_space(1))) v4i*)(uintptr_t)g,
        (__attribute__((address_space(3))) v4i*)l,
        0, 0);
#else
    *(float4*)l = *(const float4*)g;
#endif
}
__device__ inline void wait_async0() {
#if defined(HAVE_ASYNC_CP) && defined(HAVE_ASYNC_WAIT)
    __builtin_amdgcn_s_wait_asynccnt(0);
#endif
}

#if defined(HAVE_TDM)
__device__ inline void tdm_load(v4u g0, v8i g1, v4i g2, v4i g3) {
#if __clang_major__ >= 23
    __builtin_amdgcn_tensor_load_to_lds(g0, g1, g2, g3, (v8i)0, 0);
#else
    __builtin_amdgcn_tensor_load_to_lds(g0, g1, g2, g3, 0);
#endif
}
#endif

__device__ inline int pixel_index(float lx, float ly, int Hs, int Ws) {
    lx = fminf(fmaxf(lx, -1.f), 1.f);
    ly = fminf(fmaxf(ly, -1.f), 1.f);
    lx = (lx + 1.f) * 0.5f;
    ly = (ly + 1.f) * 0.5f;
    int px = (int)rintf(lx * (float)(Ws - 1));
    int py = (int)rintf(ly * (float)(Hs - 1));
    px = min(max(px, 0), Ws - 1);
    py = min(max(py, 0), Hs - 1);
    return py * Ws + px;
}

// ---------------- token2map ------------------------------------------------
__global__ __launch_bounds__(256) void t2m_count(const float* __restrict__ loc,
                                                 float* __restrict__ cnt) {
    int i = blockIdx.x * 256 + threadIdx.x;       // over B*N0
    if (i >= B_ * N0_) return;
    int b = i / N0_;
    int p = pixel_index(loc[2 * i], loc[2 * i + 1], HH, WW);
    atomicAdd(&cnt[b * HH * WW + p], 1.f);
}

__global__ __launch_bounds__(256) void t2m_scatter(const float* __restrict__ x,
                                                   const float* __restrict__ loc,
                                                   const int* __restrict__ idx_agg,
                                                   const float* __restrict__ cnt,
                                                   float* __restrict__ xmap) {
    int i = blockIdx.x * 256 + threadIdx.x;       // over B*N0*16 (4 ch each)
    if (i >= B_ * N0_ * 16) return;
    int bn0 = i >> 4;
    int c   = (i & 15) * 4;
    int b   = bn0 / N0_;
    int p   = pixel_index(loc[2 * bn0], loc[2 * bn0 + 1], HH, WW);
    size_t pf = (size_t)b * HH * WW + p;
    float val = 1.f / (cnt[pf] + EPSF);
    int tok = idx_agg[bn0];
    const float* src = x + ((size_t)b * N_TOK + tok) * CIN + c;
    float* dst = xmap + pf * CIN + c;
    atomicAdd(dst + 0, src[0] * val);
    atomicAdd(dst + 1, src[1] * val);
    atomicAdd(dst + 2, src[2] * val);
    atomicAdd(dst + 3, src[3] * val);
}

// ---------------- packing (f32 -> bf16 pair dwords) ------------------------
__global__ __launch_bounds__(256) void pack_pairs(const float* __restrict__ src,
                                                  unsigned* __restrict__ dst, int total) {
    int i = blockIdx.x * 256 + threadIdx.x;
    if (i >= total) return;
    dst[i] = packbf(src[2 * i], src[2 * i + 1]);
}

__global__ __launch_bounds__(256) void pack_convw(const float* __restrict__ cw,
                                                  unsigned* __restrict__ w_pk) {
    int i = blockIdx.x * 256 + threadIdx.x;
    if (i >= (KCONV / 2) * COUT) return;
    int k2 = i / COUT, o = i % COUT;
    float vals[2];
    #pragma unroll
    for (int j = 0; j < 2; ++j) {
        int k = 2 * k2 + j;
        int kpos = k >> 6, ci = k & 63;
        int ky = kpos / 3, kx = kpos % 3;
        vals[j] = cw[(((size_t)o * CIN + ci) * 3 + ky) * 3 + kx];
    }
    w_pk[i] = packbf(vals[0], vals[1]);
}

__global__ __launch_bounds__(256) void pack_skipw(const float* __restrict__ sw,
                                                  unsigned* __restrict__ w_pk) {
    int i = blockIdx.x * 256 + threadIdx.x;
    if (i >= (CIN / 2) * COUT) return;
    int c2 = i / COUT, o = i % COUT;
    w_pk[i] = packbf(sw[(size_t)o * CIN + 2 * c2], sw[(size_t)o * CIN + 2 * c2 + 1]);
}

// ---------------- conv as implicit GEMM with bf16 WMMA ---------------------
// M = B*H2*W2, N = 128, K = 576. One wave -> 16x16 tile; 8 waves cover 128
// cols; B-panels (16 k-pairs x 128 cols) double-buffered in LDS via async DMA.
__global__ __launch_bounds__(256) void conv_wmma(const unsigned* __restrict__ xm_pk,
                                                 const unsigned* __restrict__ w_pk,
                                                 const float* __restrict__ bias,
                                                 float* __restrict__ y) {
    __shared__ unsigned bpan[2][16 * COUT];

    const int lane = threadIdx.x & 31;
    const int wave = threadIdx.x >> 5;
    const int m0   = blockIdx.x * 16;
    const int nb   = wave * 16;
    const int half = lane >> 4;
    const int l16  = lane & 15;

    const int m   = m0 + l16;
    const int b   = m / (H2 * W2);
    const int rem = m % (H2 * W2);
    const int oy  = rem / W2;
    const int ox  = rem % W2;

    // stage first B panel
    {
        const unsigned* g = w_pk;
        for (int c = threadIdx.x; c < (16 * COUT) / 4; c += 256)
            async_cp16(g + 4 * c, &bpan[0][4 * c]);
        wait_async0();
    }
    __syncthreads();

    CFr acc;
    #pragma unroll
    for (int i = 0; i < 8; ++i) acc.f[i] = 0.f;

    int buf = 0;
    for (int kb = 0; kb < KCONV; kb += 32) {
        // stage next panel while computing this one
        if (kb + 32 < KCONV) {
            const unsigned* g = w_pk + (size_t)((kb + 32) >> 1) * COUT;
            for (int c = threadIdx.x; c < (16 * COUT) / 4; c += 256)
                async_cp16(g + 4 * c, &bpan[buf ^ 1][4 * c]);
        }

        // A addressing: kpos is constant within a 32-wide k-step
        const int kpos = kb >> 6;
        const int ky = kpos / 3, kx = kpos % 3;
        const int iy = oy * 2 - 1 + ky;
        const int ix = ox * 2 - 1 + kx;
        const bool inb = ((unsigned)iy < HH) && ((unsigned)ix < WW);
        const unsigned* ap =
            xm_pk + (((size_t)b * HH + iy) * WW + ix) * (CIN / 2) + ((kb & 63) >> 1);

        ABf a, bm;
        #pragma unroll
        for (int v = 0; v < 8; ++v) {
            int k2 = kpair_of(v, half);
            a.u[v]  = inb ? ap[k2] : 0u;
            bm.u[v] = bpan[buf][k2 * COUT + nb + l16];
        }
        acc.v = __builtin_amdgcn_wmma_f32_16x16x32_bf16(
            false, a.v, false, bm.v, (short)0, acc.v, false, false);

        wait_async0();
        __syncthreads();
        buf ^= 1;
    }

    const int n  = nb + l16;
    const float bv = bias[n];
    #pragma unroll
    for (int v = 0; v < 8; ++v) {
        int mr = m0 + v + 8 * half;             // C/D layout: lanes 0-15 M=v, 16-31 M=v+8
        y[(size_t)mr * COUT + n] = acc.f[v] + bv;
    }
}

// ---------------- map2token ------------------------------------------------
__global__ __launch_bounds__(256) void m2t_wsum(const int* __restrict__ idx_agg,
                                                const float* __restrict__ agg_w,
                                                float* __restrict__ wsum) {
    int i = blockIdx.x * 256 + threadIdx.x;       // over B*N0
    if (i >= B_ * N0_) return;
    int b = i / N0_;
    atomicAdd(&wsum[(size_t)b * N_TOK + idx_agg[i]], agg_w[i]);
}

__global__ __launch_bounds__(256) void m2t_scatter(const float* __restrict__ loc,
                                                   const int* __restrict__ idx_agg,
                                                   const float* __restrict__ agg_w,
                                                   const float* __restrict__ wsum,
                                                   const float* __restrict__ ymap,
                                                   float* __restrict__ x_tok) {
    int i = blockIdx.x * 256 + threadIdx.x;       // over B*N0*32 (4 ch each)
    if (i >= B_ * N0_ * 32) return;
    int bn0 = i >> 5;
    int c   = (i & 31) * 4;
    int b   = bn0 / N0_;
    int p   = pixel_index(loc[2 * bn0], loc[2 * bn0 + 1], H2, W2);
    int tok = idx_agg[bn0];
    float w   = agg_w[bn0];
    float val = w / (wsum[(size_t)b * N_TOK + tok] + EPSF);
    const float* src = ymap + ((size_t)b * H2 * W2 + p) * COUT + c;
    float* dst = x_tok + ((size_t)b * N_TOK + tok) * COUT + c;
    atomicAdd(dst + 0, src[0] * val);
    atomicAdd(dst + 1, src[1] * val);
    atomicAdd(dst + 2, src[2] * val);
    atomicAdd(dst + 3, src[3] * val);
}

// ---------------- skip GEMM with WMMA: x_new = x_tok + x @ skip_w^T --------
// Weight panel (32 k-pairs x 128 cols, 16 KB) staged once via TDM when the
// toolchain exposes tensor_load_to_lds, else via async/sync copy.
__global__ __launch_bounds__(256) void skip_wmma(const unsigned* __restrict__ x_pk,
                                                 const unsigned* __restrict__ w_pk,
                                                 const float* __restrict__ x_tok,
                                                 float* __restrict__ x_new) {
    __shared__ unsigned swl[(CIN / 2) * COUT];

#if defined(HAVE_TDM)
    if (threadIdx.x < 32) {   // wave 0 issues the TDM descriptor
        unsigned long long ga = (unsigned long long)(uintptr_t)w_pk;
        unsigned lds_off =
            (unsigned)(size_t)(__attribute__((address_space(3))) unsigned*)(void*)&swl[0];
        v4u g0;
        g0[0] = 1u;                                           // count=1, user D#
        g0[1] = lds_off;                                      // lds_addr
        g0[2] = (unsigned)(ga & 0xFFFFFFFFull);               // global_addr lo
        g0[3] = (unsigned)((ga >> 32) & 0x01FFFFFFull) | (2u << 30);  // hi | type=2
        v8i g1;
        g1[0] = 0x20000;                    // data_size = 4B (code 2)
        g1[1] = (int)(128u << 16);          // tensor_dim0 = 128 dwords
        g1[2] = (int)((unsigned)(CIN / 2) << 16);  // tensor_dim1 = 32
        g1[3] = (int)(128u << 16);          // tile_dim0 = 128
        g1[4] = (CIN / 2);                  // tile_dim1 = 32
        g1[5] = 128;                        // tensor_dim0_stride = 128
        g1[6] = 0;
        g1[7] = 0;
        v4i gz = (v4i)0;
        tdm_load(g0, g1, gz, gz);
        __builtin_amdgcn_s_wait_tensorcnt(0);
    }
#else
    for (int c = threadIdx.x; c < ((CIN / 2) * COUT) / 4; c += 256)
        async_cp16(w_pk + 4 * c, &swl[4 * c]);
    wait_async0();
#endif
    __syncthreads();

    const int lane = threadIdx.x & 31;
    const int wave = threadIdx.x >> 5;
    const int m0   = blockIdx.x * 16;
    const int nb   = wave * 16;
    const int half = lane >> 4;
    const int l16  = lane & 15;
    const int m    = m0 + l16;

    CFr acc;
    #pragma unroll
    for (int i = 0; i < 8; ++i) acc.f[i] = 0.f;

    #pragma unroll
    for (int kb = 0; kb < CIN; kb += 32) {
        ABf a, bm;
        #pragma unroll
        for (int v = 0; v < 8; ++v) {
            int k2 = kpair_of(v, half);
            a.u[v]  = x_pk[(size_t)m * (CIN / 2) + (kb >> 1) + k2];
            bm.u[v] = swl[((kb >> 1) + k2) * COUT + nb + l16];
        }
        acc.v = __builtin_amdgcn_wmma_f32_16x16x32_bf16(
            false, a.v, false, bm.v, (short)0, acc.v, false, false);
    }

    const int n = nb + l16;
    #pragma unroll
    for (int v = 0; v < 8; ++v) {
        int mr = m0 + v + 8 * half;
        size_t o = (size_t)mr * COUT + n;
        x_new[o] = acc.f[v] + x_tok[o];
    }
}

// ---------------- farthest point sampling (1 block / batch) ----------------
__global__ __launch_bounds__(256) void fps_kernel(const float* __restrict__ xnew,
                                                  int* __restrict__ index_down) {
    __shared__ float dist[N_TOK];
    __shared__ float cent[COUT];
    __shared__ float redv[256];
    __shared__ int   redi[256];
    __shared__ int   s_far;
    const int b = blockIdx.x;
    const int tid = threadIdx.x;
    for (int n = tid; n < N_TOK; n += 256) dist[n] = 1e10f;
    if (tid == 0) s_far = 0;
    __syncthreads();
    for (int s = 0; s < NS; ++s) {
        const int far = s_far;
        if (tid == 0) index_down[b * NS + s] = far;
        if (tid < COUT) cent[tid] = xnew[((size_t)b * N_TOK + far) * COUT + tid];
        __syncthreads();
        float best = -1.f; int bi = 0;
        for (int n = tid; n < N_TOK; n += 256) {
            const float4* xp = (const float4*)(xnew + ((size_t)b * N_TOK + n) * COUT);
            float d = 0.f;
            #pragma unroll
            for (int c4 = 0; c4 < COUT / 4; ++c4) {
                float4 xv = xp[c4];
                const float4 cv = *(const float4*)(cent + 4 * c4);
                float d0 = xv.x - cv.x, d1 = xv.y - cv.y;
                float d2 = xv.z - cv.z, d3 = xv.w - cv.w;
                d += d0 * d0 + d1 * d1 + d2 * d2 + d3 * d3;
            }
            float nd = fminf(dist[n], d);
            dist[n] = nd;
            if (nd > best) { best = nd; bi = n; }
        }
        redv[tid] = best; redi[tid] = bi;
        __syncthreads();
        for (int off = 128; off > 0; off >>= 1) {
            if (tid < off && redv[tid + off] > redv[tid]) {
                redv[tid] = redv[tid + off];
                redi[tid] = redi[tid + off];
            }
            __syncthreads();
        }
        if (tid == 0) s_far = redi[0];
        __syncthreads();
    }
}

// ---------------- gather x_down0 = x_new[index_down] -----------------------
__global__ __launch_bounds__(256) void gather_down(const float* __restrict__ xnew,
                                                   const int* __restrict__ index_down,
                                                   float* __restrict__ xdown0) {
    int i = blockIdx.x * 256 + threadIdx.x;       // over B*NS*32 float4s
    if (i >= B_ * NS * (COUT / 4)) return;
    int bs = i / (COUT / 4);
    int c4 = i % (COUT / 4);
    int b  = bs / NS;
    int src = index_down[bs];
    ((float4*)xdown0)[(size_t)bs * (COUT / 4) + c4] =
        ((const float4*)(xnew + ((size_t)b * N_TOK + src) * COUT))[c4];
}

// ---------------- nearest-centroid assignment ------------------------------
// Centroid tiles double-buffered in LDS via async DMA.
__global__ __launch_bounds__(256) void assign_kernel(const float* __restrict__ xnew,
                                                     const float* __restrict__ xdown,
                                                     int* __restrict__ idx_t) {
    __shared__ float tile[2][16 * COUT];

    const int bpb = (N_TOK + 255) / 256;          // blocks per batch
    const int b   = blockIdx.x / bpb;
    const int n   = (blockIdx.x % bpb) * 256 + threadIdx.x;
    const bool act = (n < N_TOK);

    float4 xr[COUT / 4];
    const float4* xp = (const float4*)(xnew + ((size_t)b * N_TOK + (act ? n : 0)) * COUT);
    #pragma unroll
    for (int i = 0; i < COUT / 4; ++i) xr[i] = xp[i];

    // stage first tile
    {
        const float* g = xdown + (size_t)b * NS * COUT;
        for (int i = threadIdx.x; i < 16 * COUT / 4; i += 256)
            async_cp16(g + 4 * i, &tile[0][4 * i]);
        wait_async0();
    }
    __syncthreads();

    float best = 3.4e38f; int bi = 0;
    int buf = 0;
    for (int s0 = 0; s0 < NS; s0 += 16) {
        if (s0 + 16 < NS) {
            const float* g = xdown + ((size_t)b * NS + s0 + 16) * COUT;
            for (int i = threadIdx.x; i < 16 * COUT / 4; i += 256)
                async_cp16(g + 4 * i, &tile[buf ^ 1][4 * i]);
        }
        #pragma unroll 4
        for (int s = 0; s < 16; ++s) {
            const float4* cp = (const float4*)(&tile[buf][s * COUT]);
            float d = 0.f;
            #pragma unroll
            for (int i = 0; i < COUT / 4; ++i) {
                float4 c = cp[i];
                float d0 = xr[i].x - c.x, d1 = xr[i].y - c.y;
                float d2 = xr[i].z - c.z, d3 = xr[i].w - c.w;
                d += d0 * d0 + d1 * d1 + d2 * d2 + d3 * d3;
            }
            if (d < best) { best = d; bi = s0 + s; }
        }
        wait_async0();
        __syncthreads();
        buf ^= 1;
    }
    if (act) idx_t[(size_t)b * N_TOK + n] = bi;
}

__global__ __launch_bounds__(256) void override_idx(const int* __restrict__ index_down,
                                                    int* __restrict__ idx_t) {
    int i = blockIdx.x * 256 + threadIdx.x;       // over B*NS
    if (i >= B_ * NS) return;
    int b = i / NS, s = i % NS;
    idx_t[(size_t)b * N_TOK + index_down[i]] = s;
}

__global__ __launch_bounds__(256) void count_allw(const int* __restrict__ idx_t,
                                                  float* __restrict__ allw) {
    int i = blockIdx.x * 256 + threadIdx.x;       // over B*N
    if (i >= B_ * N_TOK) return;
    int b = i / N_TOK;
    atomicAdd(&allw[(size_t)b * NS + idx_t[i]], 1.f);
}

__global__ __launch_bounds__(256) void merge_scatter(const float* __restrict__ xnew,
                                                     const int* __restrict__ idx_t,
                                                     const float* __restrict__ allw,
                                                     float* __restrict__ out_xdown) {
    int i = blockIdx.x * 256 + threadIdx.x;       // over B*N*32 (4 ch each)
    if (i >= B_ * N_TOK * 32) return;
    int bn = i >> 5;
    int c  = (i & 31) * 4;
    int b  = bn / N_TOK;
    int t  = idx_t[bn];
    float nw = 1.f / (allw[(size_t)b * NS + t] + EPSF);
    const float* src = xnew + (size_t)bn * COUT + c;
    float* dst = out_xdown + ((size_t)b * NS + t) * COUT + c;
    atomicAdd(dst + 0, src[0] * nw);
    atomicAdd(dst + 1, src[1] * nw);
    atomicAdd(dst + 2, src[2] * nw);
    atomicAdd(dst + 3, src[3] * nw);
}

__global__ __launch_bounds__(256) void out_idx_weight(const int* __restrict__ idx_agg,
                                                      const float* __restrict__ agg_w,
                                                      const int* __restrict__ idx_t,
                                                      const float* __restrict__ allw,
                                                      float* __restrict__ out_idx,
                                                      float* __restrict__ aw_tmp,
                                                      unsigned* __restrict__ maxw) {
    int i = blockIdx.x * 256 + threadIdx.x;       // over B*N0
    if (i >= B_ * N0_) return;
    int b = i / N0_;
    int t = idx_t[(size_t)b * N_TOK + idx_agg[i]];
    out_idx[i] = (float)t;
    float wt = 1.f / (allw[(size_t)b * NS + t] + EPSF);
    float aw = agg_w[i] * wt;
    aw_tmp[i] = aw;
    atomicMax(&maxw[b], __float_as_uint(aw));     // aw > 0 -> uint order == float order
}

__global__ __launch_bounds__(256) void norm_aw(const float* __restrict__ aw_tmp,
                                               const unsigned* __restrict__ maxw,
                                               float* __restrict__ out_aw) {
    int i = blockIdx.x * 256 + threadIdx.x;       // over B*N0
    if (i >= B_ * N0_) return;
    int b = i / N0_;
    out_aw[i] = aw_tmp[i] / __uint_as_float(maxw[b]);
}

// ---------------- host side ------------------------------------------------
static inline int cdiv_i(long long a, long long b) { return (int)((a + b - 1) / b); }

extern "C" void kernel_launch(void* const* d_in, const int* in_sizes, int n_in,
                              void* d_out, int out_size, void* d_ws, size_t ws_size,
                              hipStream_t stream) {
    (void)in_sizes; (void)n_in; (void)out_size; (void)ws_size;
    const float* x       = (const float*)d_in[0];
    const float* loc     = (const float*)d_in[1];
    const int*   idx_agg = (const int*)d_in[2];
    const float* agg_w   = (const float*)d_in[3];
    const float* conv_w  = (const float*)d_in[4];
    const float* conv_b  = (const float*)d_in[5];
    const float* skip_w  = (const float*)d_in[6];

    char* wp = (char*)d_ws;
    auto alloc = [&](size_t bytes) -> void* {
        void* p = wp; wp += (bytes + 255) & ~(size_t)255; return p;
    };
    float*    cnt        = (float*)   alloc((size_t)B_ * HH * WW * 4);
    float*    xmap       = (float*)   alloc((size_t)B_ * HH * WW * CIN * 4);
    unsigned* xm_pk      = (unsigned*)alloc((size_t)B_ * HH * WW * (CIN / 2) * 4);
    unsigned* cw_pk      = (unsigned*)alloc((size_t)(KCONV / 2) * COUT * 4);
    unsigned* sw_pk      = (unsigned*)alloc((size_t)(CIN / 2) * COUT * 4);
    unsigned* x_pk       = (unsigned*)alloc((size_t)B_ * N_TOK * (CIN / 2) * 4);
    float*    ymap       = (float*)   alloc((size_t)B_ * H2 * W2 * COUT * 4);
    float*    wsum       = (float*)   alloc((size_t)B_ * N_TOK * 4);
    float*    x_tok      = (float*)   alloc((size_t)B_ * N_TOK * COUT * 4);
    float*    x_new      = (float*)   alloc((size_t)B_ * N_TOK * COUT * 4);
    int*      index_down = (int*)     alloc((size_t)B_ * NS * 4);
    float*    xdown0     = (float*)   alloc((size_t)B_ * NS * COUT * 4);
    int*      idx_t      = (int*)     alloc((size_t)B_ * N_TOK * 4);
    float*    allw       = (float*)   alloc((size_t)B_ * NS * 4);
    float*    aw_tmp     = (float*)   alloc((size_t)B_ * N0_ * 4);
    unsigned* maxw       = (unsigned*)alloc((size_t)B_ * 4);

    float* out       = (float*)d_out;
    float* out_xdown = out;                                   // B*NS*COUT
    float* out_idx   = out + (size_t)B_ * NS * COUT;          // B*N0
    float* out_aw    = out_idx + (size_t)B_ * N0_;            // B*N0

    (void)hipMemsetAsync(cnt,       0, (size_t)B_ * HH * WW * 4, stream);
    (void)hipMemsetAsync(xmap,      0, (size_t)B_ * HH * WW * CIN * 4, stream);
    (void)hipMemsetAsync(wsum,      0, (size_t)B_ * N_TOK * 4, stream);
    (void)hipMemsetAsync(x_tok,     0, (size_t)B_ * N_TOK * COUT * 4, stream);
    (void)hipMemsetAsync(allw,      0, (size_t)B_ * NS * 4, stream);
    (void)hipMemsetAsync(maxw,      0, (size_t)B_ * 4, stream);
    (void)hipMemsetAsync(out_xdown, 0, (size_t)B_ * NS * COUT * 4, stream);

    const dim3 blk(256);
    t2m_count   <<<cdiv_i((long long)B_ * N0_, 256), blk, 0, stream>>>(loc, cnt);
    t2m_scatter <<<cdiv_i((long long)B_ * N0_ * 16, 256), blk, 0, stream>>>(x, loc, idx_agg, cnt, xmap);
    pack_pairs  <<<cdiv_i((long long)B_ * HH * WW * (CIN / 2), 256), blk, 0, stream>>>(
        xmap, xm_pk, B_ * HH * WW * (CIN / 2));
    pack_convw  <<<cdiv_i((KCONV / 2) * COUT, 256), blk, 0, stream>>>(conv_w, cw_pk);
    conv_wmma   <<<dim3((B_ * H2 * W2) / 16), blk, 0, stream>>>(xm_pk, cw_pk, conv_b, ymap);
    m2t_wsum    <<<cdiv_i((long long)B_ * N0_, 256), blk, 0, stream>>>(idx_agg, agg_w, wsum);
    m2t_scatter <<<cdiv_i((long long)B_ * N0_ * 32, 256), blk, 0, stream>>>(
        loc, idx_agg, agg_w, wsum, ymap, x_tok);
    pack_pairs  <<<cdiv_i((long long)B_ * N_TOK * (CIN / 2), 256), blk, 0, stream>>>(
        x, x_pk, B_ * N_TOK * (CIN / 2));
    pack_skipw  <<<cdiv_i((CIN / 2) * COUT, 256), blk, 0, stream>>>(skip_w, sw_pk);
    skip_wmma   <<<dim3((B_ * N_TOK) / 16), blk, 0, stream>>>(x_pk, sw_pk, x_tok, x_new);
    fps_kernel  <<<dim3(B_), blk, 0, stream>>>(x_new, index_down);
    gather_down <<<cdiv_i((long long)B_ * NS * (COUT / 4), 256), blk, 0, stream>>>(
        x_new, index_down, xdown0);
    assign_kernel<<<dim3(B_ * ((N_TOK + 255) / 256)), blk, 0, stream>>>(x_new, xdown0, idx_t);
    override_idx<<<cdiv_i((long long)B_ * NS, 256), blk, 0, stream>>>(index_down, idx_t);
    count_allw  <<<cdiv_i((long long)B_ * N_TOK, 256), blk, 0, stream>>>(idx_t, allw);
    merge_scatter<<<cdiv_i((long long)B_ * N_TOK * 32, 256), blk, 0, stream>>>(
        x_new, idx_t, allw, out_xdown);
    out_idx_weight<<<cdiv_i((long long)B_ * N0_, 256), blk, 0, stream>>>(
        idx_agg, agg_w, idx_t, allw, out_idx, aw_tmp, maxw);
    norm_aw     <<<cdiv_i((long long)B_ * N0_, 256), blk, 0, stream>>>(aw_tmp, maxw, out_aw);
}
